// QGNN_28441273434751
// MI455X (gfx1250) — matrix operand order repaired and checked by reading, not compiled, not run on hardware
//
#include <hip/hip_runtime.h>

#define NNODES 50000
#define NEDGES 1600000

typedef __attribute__((ext_vector_type(16))) __bf16 bf16x16;
typedef __attribute__((ext_vector_type(2)))  __bf16 bf16x2;
typedef __attribute__((ext_vector_type(8)))  float  f32x8;
typedef __attribute__((ext_vector_type(2)))  float  f32x2;
typedef __attribute__((ext_vector_type(4)))  unsigned int u32x4;
typedef __attribute__((ext_vector_type(4)))  float  f32x4;

union ABu { bf16x16 v; u32x4 q[2]; };

// hardware f32 -> bf16 (v_cvt_pk_bf16_f32 on gfx1250)
__device__ __forceinline__ unsigned int pk2(float lo, float hi) {
  f32x2 f = {lo, hi};
  bf16x2 b = __builtin_convertvector(f, bf16x2);
  return __builtin_bit_cast(unsigned int, b);
}
__device__ __forceinline__ unsigned short bf1(float f) {
  __bf16 b = (__bf16)f;
  return __builtin_bit_cast(unsigned short, b);
}

// ---------------- weight prep: fp32 -> zero-padded bf16 ----------------
// W1b: [3][64][160], W2b: [3][128][192]
__global__ void qgnn_prep_w1(const float* __restrict__ W1_0,
                             const float* __restrict__ W1_s,
                             unsigned short* __restrict__ W1b) {
  int i = blockIdx.x * blockDim.x + threadIdx.x;
  const int total = 3 * 64 * 160;
  if (i >= total) return;
  int l = i / (64 * 160), r = i % (64 * 160), o = r / 160, k = r % 160;
  float v = 0.f;
  if (l == 0) { if (k < 35)  v = W1_0[o * 35 + k]; }
  else        { if (k < 131) v = W1_s[((l - 1) * 64 + o) * 131 + k]; }
  W1b[i] = bf1(v);
}

__global__ void qgnn_prep_w2(const float* __restrict__ W2_0,
                             const float* __restrict__ W2_s,
                             unsigned short* __restrict__ W2b) {
  int i = blockIdx.x * blockDim.x + threadIdx.x;
  const int total = 3 * 128 * 192;
  if (i >= total) return;
  int l = i / (128 * 192), r = i % (128 * 192), o = r / 192, k = r % 192;
  float v = 0.f;
  if (l == 0) { if (k < 96) v = W2_0[o * 96 + k]; }
  else        { v = W2_s[((l - 1) * 128 + o) * 192 + k]; }
  W2b[i] = bf1(v);
}

// h0 = bf16(emb[gate_type])  [N,32]
__global__ void qgnn_gather_h0(const int* __restrict__ gt,
                               const float* __restrict__ emb,
                               unsigned short* __restrict__ hb0) {
  int i = blockIdx.x * blockDim.x + threadIdx.x;
  if (i >= NNODES * 32) return;
  int node = i >> 5, k = i & 31;
  hb0[i] = bf1(emb[gt[node] * 32 + k]);
}

__global__ void qgnn_zero_f32(float* __restrict__ p, int n) {
  int i = blockIdx.x * blockDim.x + threadIdx.x;
  if (i < n) p[i] = 0.f;
}

__global__ void qgnn_deg_count(const int* __restrict__ dst, int* __restrict__ deg) {
  int i = blockIdx.x * blockDim.x + threadIdx.x;
  if (i < NEDGES) atomicAdd(&deg[dst[i]], 1);
}

__global__ void qgnn_make_recip(const int* __restrict__ deg, float* __restrict__ recip) {
  int i = blockIdx.x * blockDim.x + threadIdx.x;
  if (i < NNODES) recip[i] = 1.0f / fmaxf((float)deg[i], 1.0f);
}

// ---------------- edge MLP: tmp = leaky_relu(cat(h[src],w) @ W1^T); atomic scatter to sums[dst]
// K layout: [0,F) = h features (uniform loop), [F,F+3) = w (single branchless tail iter, Kc=F+32)
__global__ __launch_bounds__(256) void qgnn_edge_mlp(
    const unsigned short* __restrict__ hb, const int* __restrict__ src,
    const int* __restrict__ dst, const float* __restrict__ w,
    const unsigned short* __restrict__ W1b, float* __restrict__ sums,
    int F) {
  __shared__ __align__(16) unsigned short sW1[64 * 160];
  {
    u32x4* dq = (u32x4*)sW1;
    const u32x4* sq = (const u32x4*)W1b;
    for (int i = threadIdx.x; i < (64 * 160 / 8); i += 256) dq[i] = sq[i];
  }
  __syncthreads();

  const int wave = threadIdx.x >> 5;
  const int lane = threadIdx.x & 31;
  const int col  = lane & 15;
  const int half = lane >> 4;
  const int eBase = (blockIdx.x * 8 + wave) * 16;
  if (eBase >= NEDGES) return;  // wave-uniform

  int e = eBase + col; if (e >= NEDGES) e = NEDGES - 1;
  const int s = src[e];
  const u32x4* hrow = (const u32x4*)(hb + (size_t)s * (size_t)F);
  const float* wp = w + (size_t)e * 3;
  const float w0 = wp[0], w1 = wp[1], w2 = wp[2];

  int dmr[8];
#pragma unroll
  for (int v = 0; v < 8; ++v) {
    int em = eBase + half * 8 + v;
    dmr[v] = (em < NEDGES) ? dst[em] : -1;
  }

  f32x8 acc[4] = {};

  // phase 1: pure h-feature K range [0, F)
  for (int k0 = 0; k0 < F; k0 += 32) {
    ABu A;
    A.q[0] = hrow[(k0 + half * 16) >> 3];
    A.q[1] = hrow[(k0 + half * 16 + 8) >> 3];
    const int kb = k0 + half * 16;
#pragma unroll
    for (int nt = 0; nt < 4; ++nt) {
      ABu B;
      const u32x4* bp = (const u32x4*)(sW1 + (nt * 16 + col) * 160 + kb);
      B.q[0] = bp[0]; B.q[1] = bp[1];
      acc[nt] = __builtin_amdgcn_wmma_f32_16x16x32_bf16(
          false, A.v, false, B.v, (short)0, acc[nt], false, false);
    }
  }

  // phase 2: tail iteration at k0 = F holding [w0,w1,w2,0,...] (branchless select)
  {
    u32x4 qz = (u32x4)(0u);
    u32x4 qw = qz;
    qw.x = pk2(w0, w1);
    qw.y = pk2(w2, 0.f);
    ABu A;
    A.q[0] = half ? qz : qw;  // w chunk lives at K=[F,F+8), i.e. half 0, chunk 0
    A.q[1] = qz;
    const int kb = F + half * 16;
#pragma unroll
    for (int nt = 0; nt < 4; ++nt) {
      ABu B;
      const u32x4* bp = (const u32x4*)(sW1 + (nt * 16 + col) * 160 + kb);
      B.q[0] = bp[0]; B.q[1] = bp[1];
      acc[nt] = __builtin_amdgcn_wmma_f32_16x16x32_bf16(
          false, A.v, false, B.v, (short)0, acc[nt], false, false);
    }
  }

#pragma unroll
  for (int nt = 0; nt < 4; ++nt) {
#pragma unroll
    for (int v = 0; v < 8; ++v) {
      if (dmr[v] >= 0) {
        float x = acc[nt][v];
        x = x > 0.f ? x : 0.01f * x;  // leaky_relu
        unsafeAtomicAdd(&sums[(size_t)dmr[v] * 64 + nt * 16 + col], x);
      }
    }
  }
}

// ---------------- node MLP: out = relu(cat(h, sums*recip) @ W2^T + b2)
// K layout: [0,F) = bf16 h (uniform loop), [F,F+64) = fp32 sums*recip (uniform loop)
__global__ __launch_bounds__(256) void qgnn_node_mlp(
    const unsigned short* __restrict__ hb, const float* __restrict__ sums,
    const float* __restrict__ recip, const unsigned short* __restrict__ W2b,
    const float* __restrict__ b2, float* __restrict__ outF,
    unsigned short* __restrict__ hbOut, int F) {
  __shared__ __align__(16) unsigned short sW2[128 * 192];
  {
    u32x4* dq = (u32x4*)sW2;
    const u32x4* sq = (const u32x4*)W2b;
    for (int i = threadIdx.x; i < (128 * 192 / 8); i += 256) dq[i] = sq[i];
  }
  __syncthreads();

  const int wave = threadIdx.x >> 5;
  const int lane = threadIdx.x & 31;
  const int col  = lane & 15;
  const int half = lane >> 4;
  const int nodeBase = (blockIdx.x * 8 + wave) * 16;
  if (nodeBase >= NNODES) return;  // wave-uniform

  int nr = nodeBase + col; if (nr >= NNODES) nr = NNODES - 1;
  const u32x4* hrow = (const u32x4*)(hb + (size_t)nr * (size_t)F);
  const f32x4* srow = (const f32x4*)(sums + (size_t)nr * 64);
  const float rc = recip[nr];

  float bb[8];
#pragma unroll
  for (int nt = 0; nt < 8; ++nt) bb[nt] = b2[nt * 16 + col];

  f32x8 acc[8] = {};

  // phase 1: bf16 h features, K in [0, F)
  for (int k0 = 0; k0 < F; k0 += 32) {
    ABu A;
    A.q[0] = hrow[(k0 + half * 16) >> 3];
    A.q[1] = hrow[(k0 + half * 16 + 8) >> 3];
    const int kb = k0 + half * 16;
#pragma unroll
    for (int nt = 0; nt < 8; ++nt) {
      ABu B;
      const u32x4* bp = (const u32x4*)(sW2 + (nt * 16 + col) * 192 + kb);
      B.q[0] = bp[0]; B.q[1] = bp[1];
      acc[nt] = __builtin_amdgcn_wmma_f32_16x16x32_bf16(
          false, A.v, false, B.v, (short)0, acc[nt], false, false);
    }
  }

  // phase 2: neighbor means (fp32 sums * recip -> bf16), K in [F, F+64): exactly 2 iters
#pragma unroll
  for (int ki = 0; ki < 2; ++ki) {
    const int k0 = ki * 32;  // offset within sums region
    ABu A;
#pragma unroll
    for (int c = 0; c < 2; ++c) {
      const int k2 = k0 + half * 16 + c * 8;
      f32x4 a = srow[k2 >> 2];
      f32x4 b = srow[(k2 >> 2) + 1];
      u32x4 q;
      q.x = pk2(a.x * rc, a.y * rc);
      q.y = pk2(a.z * rc, a.w * rc);
      q.z = pk2(b.x * rc, b.y * rc);
      q.w = pk2(b.z * rc, b.w * rc);
      A.q[c] = q;
    }
    const int kb = F + k0 + half * 16;
#pragma unroll
    for (int nt = 0; nt < 8; ++nt) {
      ABu B;
      const u32x4* bp = (const u32x4*)(sW2 + (nt * 16 + col) * 192 + kb);
      B.q[0] = bp[0]; B.q[1] = bp[1];
      acc[nt] = __builtin_amdgcn_wmma_f32_16x16x32_bf16(
          false, A.v, false, B.v, (short)0, acc[nt], false, false);
    }
  }

#pragma unroll
  for (int nt = 0; nt < 8; ++nt) {
#pragma unroll
    for (int v = 0; v < 8; ++v) {
      int node = nodeBase + half * 8 + v;
      if (node < NNODES) {
        float x = fmaxf(acc[nt][v] + bb[nt], 0.f);
        int feat = nt * 16 + col;
        if (outF)  outF[(size_t)node * 128 + feat] = x;
        if (hbOut) hbOut[(size_t)node * 128 + feat] = bf1(x);
      }
    }
  }
}

extern "C" void kernel_launch(void* const* d_in, const int* in_sizes, int n_in,
                              void* d_out, int out_size, void* d_ws, size_t ws_size,
                              hipStream_t stream) {
  const int*   gate_type = (const int*)d_in[0];
  const int*   src  = (const int*)d_in[1];
  const int*   dst  = (const int*)d_in[2];
  const float* w    = (const float*)d_in[3];
  const float* emb  = (const float*)d_in[4];
  const float* W1_0 = (const float*)d_in[5];
  const float* W2_0 = (const float*)d_in[6];
  const float* b2_0 = (const float*)d_in[7];
  const float* W1_s = (const float*)d_in[8];
  const float* W2_s = (const float*)d_in[9];
  const float* b2_s = (const float*)d_in[10];
  float* out = (float*)d_out;

  // workspace carve (256B aligned)
  char* wsp = (char*)d_ws;
  auto bump = [&](size_t bytes) -> void* {
    void* p = (void*)wsp;
    wsp += (bytes + 255) & ~(size_t)255;
    return p;
  };
  float*          sums  = (float*)bump((size_t)NNODES * 64 * 4);
  int*            deg   = (int*)bump((size_t)NNODES * 4);
  float*          recip = (float*)bump((size_t)NNODES * 4);
  unsigned short* hb0   = (unsigned short*)bump((size_t)NNODES * 32 * 2);
  unsigned short* hbA   = (unsigned short*)bump((size_t)NNODES * 128 * 2);
  unsigned short* hbB   = (unsigned short*)bump((size_t)NNODES * 128 * 2);
  unsigned short* W1b   = (unsigned short*)bump((size_t)3 * 64 * 160 * 2);
  unsigned short* W2b   = (unsigned short*)bump((size_t)3 * 128 * 192 * 2);

  // one-time prep
  qgnn_prep_w1<<<(3 * 64 * 160 + 255) / 256, 256, 0, stream>>>(W1_0, W1_s, W1b);
  qgnn_prep_w2<<<(3 * 128 * 192 + 255) / 256, 256, 0, stream>>>(W2_0, W2_s, W2b);
  qgnn_gather_h0<<<(NNODES * 32 + 255) / 256, 256, 0, stream>>>(gate_type, emb, hb0);
  qgnn_zero_f32<<<(NNODES + 255) / 256, 256, 0, stream>>>((float*)deg, NNODES);
  qgnn_deg_count<<<(NEDGES + 255) / 256, 256, 0, stream>>>(dst, deg);
  qgnn_make_recip<<<(NNODES + 255) / 256, 256, 0, stream>>>(deg, recip);

  const int edgeBlocks = (NEDGES + 127) / 128;
  const int nodeBlocks = (NNODES + 127) / 128;

  for (int l = 0; l < 3; ++l) {
    const int F = (l == 0) ? 32 : 128;  // input feature width
    const unsigned short* hbIn = (l == 0) ? hb0 : ((l == 1) ? hbA : hbB);
    unsigned short* hbOut = (l == 0) ? hbA : ((l == 1) ? hbB : nullptr);
    float* outF = (l == 2) ? out : nullptr;
    const float* b2 = (l == 0) ? b2_0 : (b2_s + (size_t)(l - 1) * 128);

    qgnn_zero_f32<<<(NNODES * 64 + 255) / 256, 256, 0, stream>>>(sums, NNODES * 64);
    qgnn_edge_mlp<<<edgeBlocks, 256, 0, stream>>>(
        hbIn, src, dst, w, W1b + (size_t)l * 64 * 160, sums, F);
    qgnn_node_mlp<<<nodeBlocks, 256, 0, stream>>>(
        hbIn, sums, recip, W2b + (size_t)l * 128 * 192, b2, outF, hbOut, F);
  }
}
